// DCGNN_26517128085843
// MI455X (gfx1250) — matrix operand order repaired and checked by reading, not compiled, not run on hardware
//
#include <hip/hip_runtime.h>
#include <hip/hip_bf16.h>

typedef _Float16 half_t;
typedef __attribute__((ext_vector_type(16))) _Float16 v16h;
typedef __attribute__((ext_vector_type(8)))  float    v8f;

#define BB   16
#define NN   2048
#define KK   20
#define PP   (BB * NN)          // 32768 points
#define EE   ((size_t)PP * KK)  // 655360 edges

// ---------------------------------------------------------------------------
// kNN in 3-D position space. One block per (batch, 256-query chunk); whole
// batch point cloud (2048*3 f32 = 24KB) cached in LDS. Unsorted top-20 kept in
// registers (replace-worst + unrolled rescan -> no scratch spills; unsorted is
// legal because downstream max-aggregation is order-invariant).
// ---------------------------------------------------------------------------
__global__ void knn3_kernel(const float* __restrict__ pos, int* __restrict__ idx) {
    __shared__ float sp[NN * 3];
    const int b = blockIdx.x;
    const int t = threadIdx.x;
    for (int i = t; i < NN * 3; i += 256) sp[i] = pos[(size_t)b * NN * 3 + i];
    __syncthreads();
    const int q = blockIdx.y * 256 + t;
    const float qx = sp[q * 3 + 0], qy = sp[q * 3 + 1], qz = sp[q * 3 + 2];
    float bd[KK]; int bi[KK];
#pragma unroll
    for (int s = 0; s < KK; ++s) { bd[s] = 3.0e38f; bi[s] = 0; }
    float worst = 3.0e38f; int wslot = 0;
    for (int j = 0; j < NN; ++j) {
        const float dx = sp[j * 3 + 0] - qx;
        const float dy = sp[j * 3 + 1] - qy;
        const float dz = sp[j * 3 + 2] - qz;
        const float d = fmaf(dx, dx, fmaf(dy, dy, dz * dz));
        if (d < worst) {
#pragma unroll
            for (int s = 0; s < KK; ++s) if (s == wslot) { bd[s] = d; bi[s] = j; }
            worst = bd[0]; wslot = 0;
#pragma unroll
            for (int s = 1; s < KK; ++s) if (bd[s] > worst) { worst = bd[s]; wslot = s; }
        }
    }
    int* op = idx + ((size_t)b * NN + q) * KK;
#pragma unroll
    for (int s = 0; s < KK; ++s) op[s] = b * NN + bi[s];
}

// ---------------------------------------------------------------------------
// kNN in 64-D feature space (x1, f16). Candidates streamed through LDS in
// 256-row chunks (256*64*2B = 32KB).
// ---------------------------------------------------------------------------
__global__ void knn64_kernel(const half_t* __restrict__ x1, int* __restrict__ idx) {
    __shared__ __align__(16) half_t sc[256 * 64];
    const int b = blockIdx.x;
    const int t = threadIdx.x;
    const int q = blockIdx.y * 256 + t;
    float qf[64];
    const half_t* qr = x1 + ((size_t)b * NN + q) * 64;
#pragma unroll
    for (int c = 0; c < 64; ++c) qf[c] = (float)qr[c];
    float bd[KK]; int bi[KK];
#pragma unroll
    for (int s = 0; s < KK; ++s) { bd[s] = 3.0e38f; bi[s] = 0; }
    float worst = 3.0e38f; int wslot = 0;
    for (int c0 = 0; c0 < NN; c0 += 256) {
        __syncthreads();
        // vectorized row copy: 64 halves = 8 x b128
        const uint4* crow =
            reinterpret_cast<const uint4*>(x1 + ((size_t)b * NN + c0 + t) * 64);
        uint4* srow = reinterpret_cast<uint4*>(sc + t * 64);
#pragma unroll
        for (int c = 0; c < 8; ++c) srow[c] = crow[c];
        __syncthreads();
        for (int j = 0; j < 256; ++j) {
            const half_t* cr = sc + j * 64;
            float d = 0.f;
#pragma unroll
            for (int c = 0; c < 64; ++c) {
                const float df = qf[c] - (float)cr[c];
                d = fmaf(df, df, d);
            }
            if (d < worst) {
                const int cj = c0 + j;
#pragma unroll
                for (int s = 0; s < KK; ++s) if (s == wslot) { bd[s] = d; bi[s] = cj; }
                worst = bd[0]; wslot = 0;
#pragma unroll
                for (int s = 1; s < KK; ++s) if (bd[s] > worst) { worst = bd[s]; wslot = s; }
            }
        }
    }
    int* op = idx + ((size_t)b * NN + q) * KK;
#pragma unroll
    for (int s = 0; s < KK; ++s) op[s] = b * NN + bi[s];
}

// ---------------------------------------------------------------------------
// conv1 layer a: fused edge-feature gather ([x_i, x_j - x_i], 6 wide) +
// Linear(6,64) + ReLU. K=6 is too small for WMMA; scalar FMA with weights in
// LDS. One thread per edge.
// ---------------------------------------------------------------------------
__global__ void edge1_layer_a(const float* __restrict__ pos, const int* __restrict__ idx,
                              const float* __restrict__ w, const float* __restrict__ bias,
                              half_t* __restrict__ out) {
    __shared__ float sw[6 * 64];
    __shared__ float sb[64];
    const int t = threadIdx.x;
    for (int i = t; i < 6 * 64; i += 256) sw[i] = w[i];
    if (t < 64) sb[t] = bias[t];
    __syncthreads();
    const size_t e = (size_t)blockIdx.x * 256 + t;
    if (e >= EE) return;
    const size_t i = e / KK;
    const size_t j = (size_t)idx[e];
    float ef[6];
#pragma unroll
    for (int c = 0; c < 3; ++c) {
        const float xi = pos[i * 3 + c];
        ef[c] = xi;
        ef[3 + c] = pos[j * 3 + c] - xi;
    }
    half_t* op = out + e * 64;
#pragma unroll 4
    for (int f = 0; f < 64; ++f) {
        float acc = sb[f];
#pragma unroll
        for (int c = 0; c < 6; ++c) acc = fmaf(ef[c], sw[c * 64 + f], acc);
        op[f] = (half_t)(acc > 0.f ? acc : 0.f);
    }
}

// ---------------------------------------------------------------------------
// Weight prep: W f32 [Kin,F] -> Wt f16 transposed [F,Kin]. Run once per layer
// so the GEMM's B-fragment becomes two contiguous b128 loads of f16.
// ---------------------------------------------------------------------------
__global__ void wprep_kernel(const float* __restrict__ W, half_t* __restrict__ Wt,
                             int Kin, int F) {
    const int i = blockIdx.x * 256 + threadIdx.x;
    if (i >= Kin * F) return;
    const int f = i / Kin, k = i % Kin;
    Wt[i] = (half_t)W[(size_t)k * F + f];
}

// ---------------------------------------------------------------------------
// WMMA GEMM: Y[M,F] = relu(A[M,Kin] @ W + bias), Wt = f16 [F,Kin].
// Block = 4 waves covering one 16-row M-tile x 64 output columns.
// The 16xKin A tile (contiguous in global) is staged once in LDS; fragments
// are assembled with aligned b128 loads per the CDNA5 7.12.2 layouts:
//   A lane(g,lr): rows lr, halves {k0+g*8..+7} and {k0+16+g*8..+7}
//   B lane(g,lr): col lr, halves {k0+g*8..+7} and {k0+16+g*8..+7}
//   C/D: row = 8*g + r, col = lr
// ---------------------------------------------------------------------------
__device__ __forceinline__ v8f wmma_k32(const v16h& a, const v16h& b, const v8f& c) {
    return __builtin_amdgcn_wmma_f32_16x16x32_f16(false, a, false, b, (short)0, c,
                                                  false, false);
}

union Frag32 { uint4 u[2]; v16h v; };

__global__ void wmma_gemm_bias_relu(const half_t* __restrict__ A,
                                    const half_t* __restrict__ Wt,
                                    const float* __restrict__ bias,
                                    half_t* __restrict__ Y,
                                    int Kin, int F) {
    __shared__ __align__(16) half_t sA[16 * 1024];  // up to Kin=1024
    const int t = threadIdx.x;      // 0..127
    const int wave = t >> 5;        // 0..3
    const int lane = t & 31;
    const int g = lane >> 4;
    const int lr = lane & 15;
    const int m0 = blockIdx.x * 16;
    const int n0 = blockIdx.y * 64 + wave * 16;

    // contiguous 16-row tile copy, b128 chunks
    const int nChunks = (16 * Kin) >> 3;
    const uint4* gA = reinterpret_cast<const uint4*>(A + (size_t)m0 * Kin);
    uint4* sA4 = reinterpret_cast<uint4*>(sA);
    for (int c = t; c < nChunks; c += 128) sA4[c] = gA[c];
    __syncthreads();

    v8f acc = {};
    const half_t* wrow = Wt + (size_t)(n0 + lr) * Kin;
    for (int k0 = 0; k0 < Kin; k0 += 32) {
        Frag32 a, b;
        const uint4* pa = reinterpret_cast<const uint4*>(sA + lr * Kin + k0 + g * 8);
        a.u[0] = pa[0];
        a.u[1] = pa[2];  // +16 halves
        const uint4* pb = reinterpret_cast<const uint4*>(wrow + k0 + g * 8);
        b.u[0] = pb[0];
        b.u[1] = pb[2];
        acc = wmma_k32(a.v, b.v, acc);
    }
    const int col = n0 + lr;
    const float bv = bias[col];
#pragma unroll
    for (int r = 0; r < 8; ++r) {
        const int row = m0 + g * 8 + r;
        float v = acc[r] + bv;
        v = v > 0.f ? v : 0.f;
        Y[(size_t)row * F + col] = (half_t)v;
    }
}

// ---------------------------------------------------------------------------
// conv2 WMMA GEMM with the edge-feature gather ([x1_i, x1_j - x1_i], 128 wide)
// fused into the LDS A-tile staging (avoids a 168MB edge tensor). Block =
// 8 waves covering one 16-row M-tile x all 128 output columns.
// ---------------------------------------------------------------------------
__global__ void wmma_edge_gemm2(const half_t* __restrict__ x1,
                                const int* __restrict__ idx,
                                const half_t* __restrict__ Wt,  // [128,128] f16
                                const float* __restrict__ bias,
                                half_t* __restrict__ Y) {
    const int Kin = 128, F = 128, C = 64;
    __shared__ __align__(16) half_t sA[16 * 128];
    const int t = threadIdx.x;      // 0..255
    const int wave = t >> 5;        // 0..7
    const int lane = t & 31;
    const int g = lane >> 4;
    const int lr = lane & 15;
    const int m0 = blockIdx.x * 16;
    const int n0 = wave * 16;

    // cooperative gather: thread t stages 8 consecutive features of one row
    {
        const int r = t >> 4;            // 0..15
        const int seg = t & 15;          // 16 segs x 8 halves = 128 features
        const int f0 = seg * 8;
        const size_t row = (size_t)(m0 + r);
        const size_t pi = row / KK;
        const size_t pj = (size_t)idx[row];
        const half_t* xi = x1 + pi * C;
        if (f0 < C) {
            // copy x_i chunk (aligned b128)
            *reinterpret_cast<uint4*>(sA + r * 128 + f0) =
                *reinterpret_cast<const uint4*>(xi + f0);
        } else {
            const half_t* xj = x1 + pj * C;
            const int c0 = f0 - C;
#pragma unroll
            for (int c = 0; c < 8; ++c)
                sA[r * 128 + f0 + c] =
                    (half_t)((float)xj[c0 + c] - (float)xi[c0 + c]);
        }
    }
    __syncthreads();

    v8f acc = {};
    const half_t* wrow = Wt + (size_t)(n0 + lr) * Kin;
#pragma unroll
    for (int k0 = 0; k0 < Kin; k0 += 32) {
        Frag32 a, b;
        const uint4* pa = reinterpret_cast<const uint4*>(sA + lr * Kin + k0 + g * 8);
        a.u[0] = pa[0];
        a.u[1] = pa[2];
        const uint4* pb = reinterpret_cast<const uint4*>(wrow + k0 + g * 8);
        b.u[0] = pb[0];
        b.u[1] = pb[2];
        acc = wmma_k32(a.v, b.v, acc);
    }
    const int col = n0 + lr;
    const float bv = bias[col];
#pragma unroll
    for (int r = 0; r < 8; ++r) {
        const int row = m0 + g * 8 + r;
        float v = acc[r] + bv;
        v = v > 0.f ? v : 0.f;
        Y[(size_t)row * F + col] = (half_t)v;
    }
}

// ---------------------------------------------------------------------------
// BatchNorm (training mode): stats -> finalize -> apply.
// ---------------------------------------------------------------------------
__global__ void bn_stats_kernel(const half_t* __restrict__ Y, float* __restrict__ stats,
                                int M, int F) {
    const int f = blockIdx.x * 64 + threadIdx.x;
    if (f >= F) return;
    float s = 0.f, s2 = 0.f;
    for (int r = blockIdx.y; r < M; r += gridDim.y) {
        const float v = (float)Y[(size_t)r * F + f];
        s += v;
        s2 = fmaf(v, v, s2);
    }
    atomicAdd(&stats[f], s);
    atomicAdd(&stats[F + f], s2);
}

__global__ void bn_finalize_kernel(const float* __restrict__ stats,
                                   const float* __restrict__ gamma,
                                   const float* __restrict__ beta,
                                   float* __restrict__ ss, int F, float invCnt) {
    const int f = blockIdx.x * 64 + threadIdx.x;
    if (f >= F) return;
    const float mean = stats[f] * invCnt;
    const float var = stats[F + f] * invCnt - mean * mean;  // biased var
    const float sc = gamma[f] * rsqrtf(var + 1e-5f);
    ss[f] = sc;
    ss[F + f] = beta[f] - mean * sc;
}

__global__ void bn_apply_kernel(half_t* __restrict__ Y, const float* __restrict__ ss,
                                size_t total, int F) {
    for (size_t i = (size_t)blockIdx.x * 256 + threadIdx.x; i < total;
         i += (size_t)gridDim.x * 256) {
        const int f = (int)(i % (size_t)F);
        Y[i] = (half_t)fmaf((float)Y[i], ss[f], ss[F + f]);
    }
}

// ---------------------------------------------------------------------------
// Max over the K=20 neighbor axis: [P,K,F] -> [P,F].
// ---------------------------------------------------------------------------
__global__ void max_over_k_kernel(const half_t* __restrict__ Z, half_t* __restrict__ out,
                                  size_t total, int F) {
    for (size_t i = (size_t)blockIdx.x * 256 + threadIdx.x; i < total;
         i += (size_t)gridDim.x * 256) {
        const size_t p = i / (size_t)F;
        const int f = (int)(i % (size_t)F);
        const half_t* zp = Z + (p * KK) * (size_t)F + f;
        float m = -3.0e38f;
#pragma unroll
        for (int k = 0; k < KK; ++k) m = fmaxf(m, (float)zp[(size_t)k * F]);
        out[i] = (half_t)m;
    }
}

// concat [x1 | x2] -> 192-wide rows
__global__ void concat_kernel(const half_t* __restrict__ x1, const half_t* __restrict__ x2,
                              half_t* __restrict__ out) {
    const size_t total = (size_t)PP * 192;
    for (size_t i = (size_t)blockIdx.x * 256 + threadIdx.x; i < total;
         i += (size_t)gridDim.x * 256) {
        const size_t r = i / 192;
        const int c = (int)(i % 192);
        out[i] = (c < 64) ? x1[r * 64 + c] : x2[r * 128 + (c - 64)];
    }
}

// global max pool over N: [B,N,F] -> [B,F]
__global__ void maxpool_n_kernel(const half_t* __restrict__ H, half_t* __restrict__ out,
                                 int F) {
    const int i = blockIdx.x * 256 + threadIdx.x;
    if (i >= BB * F) return;
    const int b = i / F;
    const int f = i % F;
    const half_t* hp = H + (size_t)b * NN * F + f;
    float m = -3.0e38f;
    for (int n = 0; n < NN; ++n) m = fmaxf(m, (float)hp[(size_t)n * F]);
    out[i] = (half_t)m;
}

// final Linear(256,40) + log_softmax, one block (640 threads = 20 waves)
__global__ void head_final_kernel(const half_t* __restrict__ h, const float* __restrict__ w,
                                  const float* __restrict__ b, float* __restrict__ out) {
    __shared__ float sl[BB * 40];
    __shared__ float smax[BB], slse[BB];
    const int t = threadIdx.x;
    const int row = t / 40, c = t % 40;
    float acc = b[c];
    for (int k = 0; k < 256; ++k) acc = fmaf((float)h[row * 256 + k], w[k * 40 + c], acc);
    sl[row * 40 + c] = acc;
    __syncthreads();
    if (t < BB) {
        float m = -3.0e38f;
        for (int cc = 0; cc < 40; ++cc) m = fmaxf(m, sl[t * 40 + cc]);
        float s = 0.f;
        for (int cc = 0; cc < 40; ++cc) s += __expf(sl[t * 40 + cc] - m);
        smax[t] = m;
        slse[t] = __logf(s);
    }
    __syncthreads();
    out[t] = sl[row * 40 + c] - smax[row] - slse[row];
}

// ---------------------------------------------------------------------------
extern "C" void kernel_launch(void* const* d_in, const int* in_sizes, int n_in,
                              void* d_out, int out_size, void* d_ws, size_t ws_size,
                              hipStream_t stream) {
    const float* pos = (const float*)d_in[0];
    const float* w1a = (const float*)d_in[2];
    const float* b1a = (const float*)d_in[3];
    const float* g1a = (const float*)d_in[4];
    const float* be1a = (const float*)d_in[5];
    const float* w1b = (const float*)d_in[6];
    const float* b1b = (const float*)d_in[7];
    const float* g1b = (const float*)d_in[8];
    const float* be1b = (const float*)d_in[9];
    const float* w1c = (const float*)d_in[10];
    const float* b1c = (const float*)d_in[11];
    const float* g1c = (const float*)d_in[12];
    const float* be1c = (const float*)d_in[13];
    const float* w2 = (const float*)d_in[14];
    const float* b2 = (const float*)d_in[15];
    const float* g2 = (const float*)d_in[16];
    const float* be2 = (const float*)d_in[17];
    const float* w3 = (const float*)d_in[18];
    const float* b3 = (const float*)d_in[19];
    const float* g3 = (const float*)d_in[20];
    const float* be3 = (const float*)d_in[21];
    const float* w4 = (const float*)d_in[22];
    const float* b4 = (const float*)d_in[23];
    const float* g4 = (const float*)d_in[24];
    const float* be4 = (const float*)d_in[25];
    const float* w5 = (const float*)d_in[26];
    const float* b5 = (const float*)d_in[27];
    const float* g5 = (const float*)d_in[28];
    const float* be5 = (const float*)d_in[29];
    const float* w6 = (const float*)d_in[30];
    const float* b6 = (const float*)d_in[31];

    // ---- workspace layout (bump allocator, 256B aligned) -------------------
    char* ws = (char*)d_ws;
    size_t off = 0;
    auto alloc = [&](size_t bytes) -> void* {
        void* p = ws + off;
        off += (bytes + 255) & ~(size_t)255;
        return p;
    };
    half_t* bufA = (half_t*)alloc(EE * 64 * sizeof(half_t));  // contiguous with bufB
    half_t* bufB = (half_t*)alloc(EE * 64 * sizeof(half_t));
    int* idx = (int*)alloc(EE * sizeof(int));
    half_t* x1 = (half_t*)alloc((size_t)PP * 64 * sizeof(half_t));
    half_t* x2 = (half_t*)alloc((size_t)PP * 128 * sizeof(half_t));
    half_t* h3in = (half_t*)alloc((size_t)PP * 192 * sizeof(half_t));
    float* stats = (float*)alloc(2 * 1024 * sizeof(float));
    float* ssbuf = (float*)alloc(2 * 1024 * sizeof(float));
    half_t* gpool = (half_t*)alloc((size_t)BB * 1024 * sizeof(half_t));
    half_t* h4 = (half_t*)alloc((size_t)BB * 512 * sizeof(half_t));
    half_t* h5 = (half_t*)alloc((size_t)BB * 256 * sizeof(half_t));
    half_t* wt = (half_t*)alloc((size_t)1024 * 512 * sizeof(half_t));  // max layer
    half_t* z2 = bufA;  // [E,128] overlays bufA+bufB (exactly E*128*2 bytes)
    half_t* h3 = bufA;  // [P,1024] overlays bufA (z2 dead by then)
    (void)ws_size; (void)in_sizes; (void)n_in; (void)out_size;

    auto prep_w = [&](const float* W, int Kin, int F) {
        const int total = Kin * F;
        wprep_kernel<<<(total + 255) / 256, 256, 0, stream>>>(W, wt, Kin, F);
    };
    auto run_bn = [&](half_t* Y, size_t M, int F, const float* gamma, const float* beta) {
        hipMemsetAsync(stats, 0, 2 * (size_t)F * sizeof(float), stream);
        dim3 gs((F + 63) / 64, 256);
        bn_stats_kernel<<<gs, 64, 0, stream>>>(Y, stats, (int)M, F);
        bn_finalize_kernel<<<(F + 63) / 64, 64, 0, stream>>>(stats, gamma, beta, ssbuf, F,
                                                             1.0f / (float)M);
        const size_t total = M * (size_t)F;
        size_t blk = (total + 255) / 256;
        if (blk > 8192) blk = 8192;
        bn_apply_kernel<<<(unsigned)blk, 256, 0, stream>>>(Y, ssbuf, total, F);
    };
    auto run_gemm = [&](const half_t* A, const float* W, const float* bias, half_t* Y,
                        size_t M, int Kin, int F) {
        prep_w(W, Kin, F);
        dim3 grid((unsigned)(M / 16), (unsigned)(F / 64));
        wmma_gemm_bias_relu<<<grid, 128, 0, stream>>>(A, wt, bias, Y, Kin, F);
    };

    // ---- conv1: kNN(pos) -> MLP[6,64,64,64] -> max over k ------------------
    knn3_kernel<<<dim3(BB, NN / 256), 256, 0, stream>>>(pos, idx);
    edge1_layer_a<<<(unsigned)((EE + 255) / 256), 256, 0, stream>>>(pos, idx, w1a, b1a, bufA);
    run_bn(bufA, EE, 64, g1a, be1a);
    run_gemm(bufA, w1b, b1b, bufB, EE, 64, 64);
    run_bn(bufB, EE, 64, g1b, be1b);
    run_gemm(bufB, w1c, b1c, bufA, EE, 64, 64);
    run_bn(bufA, EE, 64, g1c, be1c);
    {
        const size_t total = (size_t)PP * 64;
        max_over_k_kernel<<<(unsigned)((total + 255) / 256), 256, 0, stream>>>(bufA, x1, total, 64);
    }

    // ---- conv2: kNN(x1) -> MLP[128,128] -> max over k ----------------------
    knn64_kernel<<<dim3(BB, NN / 256), 256, 0, stream>>>(x1, idx);
    prep_w(w2, 128, 128);
    wmma_edge_gemm2<<<(unsigned)(EE / 16), 256, 0, stream>>>(x1, idx, wt, b2, z2);
    run_bn(z2, EE, 128, g2, be2);
    {
        const size_t total = (size_t)PP * 128;
        max_over_k_kernel<<<(unsigned)((total + 255) / 256), 256, 0, stream>>>(z2, x2, total, 128);
    }

    // ---- lin1: concat -> Linear(192,1024)+ReLU+BN -> global max pool -------
    {
        const size_t total = (size_t)PP * 192;
        concat_kernel<<<(unsigned)((total + 255) / 256), 256, 0, stream>>>(x1, x2, h3in);
    }
    run_gemm(h3in, w3, b3, h3, (size_t)PP, 192, 1024);
    run_bn(h3, (size_t)PP, 1024, g3, be3);
    maxpool_n_kernel<<<(BB * 1024 + 255) / 256, 256, 0, stream>>>(h3, gpool, 1024);

    // ---- head --------------------------------------------------------------
    run_gemm(gpool, w4, b4, h4, (size_t)BB, 1024, 512);
    run_bn(h4, (size_t)BB, 512, g4, be4);
    run_gemm(h4, w5, b5, h5, (size_t)BB, 512, 256);
    run_bn(h5, (size_t)BB, 256, g5, be5);
    head_final_kernel<<<1, BB * 40, 0, stream>>>(h5, w6, b6, (float*)d_out);
}
